// ScaleDotProductAttention_54975581388907
// MI455X (gfx1250) — matrix operand order repaired and checked by reading, not compiled, not run on hardware
//
#include <hip/hip_runtime.h>
#include <stdint.h>

typedef __attribute__((ext_vector_type(16))) __bf16 v16bf;
typedef __attribute__((ext_vector_type(8)))  float  v8f;

#define S_LEN 2048
#define D_DIM 128
#define QT    16          // queries per workgroup
#define SROW  2049        // padded fp32 row stride for scores (conflict-free banks)
#define QSTRIDE 130       // padded bf16 stride for Q tile
#define VSTRIDE 132       // padded fp32 stride for V tile (528B rows: 16B aligned)

#define SC_BYTES   (QT * SROW * 4)              // 131136
#define QS_BYTES   (QT * QSTRIDE * 2)           // 4160
#define VS_OFF     (SC_BYTES + QS_BYTES)        // 135296 (16B aligned)
#define VSB_BYTES  (32 * VSTRIDE * 4)           // 16896 per buffer (16B aligned)
#define SMEM_BYTES (VS_OFF + 2 * VSB_BYTES)     // 169088

__device__ __forceinline__ void async_wait_le4() {
#if __has_builtin(__builtin_amdgcn_s_wait_asynccnt)
    __builtin_amdgcn_s_wait_asynccnt(4);
#else
    asm volatile("s_wait_asynccnt 0x4" ::: "memory");
#endif
}
__device__ __forceinline__ void async_wait_le0() {
#if __has_builtin(__builtin_amdgcn_s_wait_asynccnt)
    __builtin_amdgcn_s_wait_asynccnt(0);
#else
    asm volatile("s_wait_asynccnt 0x0" ::: "memory");
#endif
}

// GVS-mode async global->LDS copy: vdst = LDS byte address, vaddr = 32-bit byte
// offset (per lane), saddr = uniform 64-bit base.  Tracked with ASYNCcnt.
__device__ __forceinline__ void async_load_b128(uint32_t lds_addr, int byte_off,
                                                const float* base) {
    asm volatile("global_load_async_to_lds_b128 %0, %1, %2"
                 :: "v"(lds_addr), "v"(byte_off), "s"(base) : "memory");
}

__launch_bounds__(256, 1)
__global__ void sdpa_fused_kernel(const float* __restrict__ q,
                                  const float* __restrict__ k,
                                  const float* __restrict__ v,
                                  float* __restrict__ out,    // [B,H,S,D]
                                  float* __restrict__ attn)   // [B,H,S,S]
{
    extern __shared__ char smem[];
    float*  sc = (float*)smem;                        // [QT][SROW] fp32 scores/P
    __bf16* qs = (__bf16*)(smem + SC_BYTES);          // [QT][QSTRIDE] bf16 Q
    float*  vs0 = (float*)(smem + VS_OFF);            // [32][VSTRIDE] fp32 V buf 0
    float*  vs1 = (float*)(smem + VS_OFF + VSB_BYTES);// [32][VSTRIDE] fp32 V buf 1
    const uint32_t vsAddr[2] = { (uint32_t)(uintptr_t)vs0, (uint32_t)(uintptr_t)vs1 };

    const int tid  = threadIdx.x;
    const int wave = tid >> 5;
    const int lane = tid & 31;
    const int half = lane >> 4;      // 0: lanes 0-15, 1: lanes 16-31
    const int l16  = lane & 15;

    const int tilesPerBH = S_LEN / QT;               // 128
    const int bh    = blockIdx.x / tilesPerBH;
    const int qBase = (blockIdx.x % tilesPerBH) * QT;

    const float* qG = q + (size_t)bh * S_LEN * D_DIM + (size_t)qBase * D_DIM;
    const float* kG = k + (size_t)bh * S_LEN * D_DIM;
    const float* vG = v + (size_t)bh * S_LEN * D_DIM;
    float* outG  = out  + (size_t)bh * S_LEN * D_DIM + (size_t)qBase * D_DIM;
    float* attnG = attn + (size_t)bh * S_LEN * S_LEN + (size_t)qBase * S_LEN;

    // ---------------- Phase 0: Q tile -> LDS (bf16), float4 loads ---
    for (int i = tid; i < QT * D_DIM / 4; i += 256) {
        int r = i >> 5, c = (i & 31) << 2;
        float4 qv = ((const float4*)qG)[i];
        __bf16* dst = qs + r * QSTRIDE + c;
        dst[0] = (__bf16)qv.x; dst[1] = (__bf16)qv.y;
        dst[2] = (__bf16)qv.z; dst[3] = (__bf16)qv.w;
    }
    __syncthreads();

    // Precompute the 4 A-fragments of Q (K-dim blocks of 32), reused for every
    // key tile.  16-bit A layout: lane half h, VGPR v(0..3): K = 8h+2v{,+1};
    // v(4..7): K = 16+8h+2(v-4){,+1};  M = lane%16.
    v16bf aQ[4];
    #pragma unroll
    for (int kb = 0; kb < 4; ++kb) {
        #pragma unroll
        for (int vg = 0; vg < 8; ++vg) {
            int kk = kb * 32 + ((vg >> 2) << 4) + half * 8 + ((vg & 3) << 1);
            aQ[kb][2 * vg]     = qs[l16 * QSTRIDE + kk];
            aQ[kb][2 * vg + 1] = qs[l16 * QSTRIDE + kk + 1];
        }
    }

    const float scale = 0.08838834764831845f;   // 1/sqrt(128)

    // ---------------- Phase 1: scores = (Q K^T) * scale -> LDS ------
    // Each wave computes 16 of the 128 (16x16) key tiles.
    for (int t = 0; t < 16; ++t) {
        const int nBase = (wave + 8 * t) * 16;   // key index base
        v8f c = {};
        #pragma unroll
        for (int kb = 0; kb < 4; ++kb) {
            const int kd = kb * 32 + half * 8;   // contraction (head-dim) index base
            // B layout (32x16 bf16): N = lane%16, K same as A layout (8h+2v{,+1}; 16+8h+2(v-4){,+1}).
            v16bf b;
            #pragma unroll
            for (int vg = 0; vg < 8; ++vg) {
                int kk = kd + ((vg >> 2) << 4) + ((vg & 3) << 1);
                b[2 * vg]     = (__bf16)kG[(size_t)(nBase + l16) * D_DIM + kk];
                b[2 * vg + 1] = (__bf16)kG[(size_t)(nBase + l16) * D_DIM + kk + 1];
            }
            c = __builtin_amdgcn_wmma_f32_16x16x32_bf16(false, aQ[kb], false, b,
                                                        (short)0, c, false, false);
        }
        // C layout: lanes 0-15 hold M=r, lanes 16-31 hold M=8+r; N = lane%16.
        #pragma unroll
        for (int r = 0; r < 8; ++r) {
            int m = half * 8 + r;
            sc[m * SROW + nBase + l16] = c[r] * scale;
        }
    }
    __syncthreads();

    // ---------------- Phase 2: row softmax, write attn (streaming) --
    #pragma unroll
    for (int rr = 0; rr < 2; ++rr) {
        const int row = wave * 2 + rr;
        float* srow = sc + row * SROW;
        float m = -INFINITY;
        for (int j = lane; j < S_LEN; j += 32) m = fmaxf(m, srow[j]);
        #pragma unroll
        for (int off = 16; off > 0; off >>= 1) m = fmaxf(m, __shfl_xor(m, off, 32));
        float sum = 0.f;
        for (int j = lane; j < S_LEN; j += 32) {
            float e = __expf(srow[j] - m);
            srow[j] = e;
            sum += e;
        }
        #pragma unroll
        for (int off = 16; off > 0; off >>= 1) sum += __shfl_xor(sum, off, 32);
        const float inv = 1.0f / sum;
        for (int j = lane; j < S_LEN; j += 32) {
            float p = srow[j] * inv;
            srow[j] = p;                                      // keep P for phase 3
            // attn is written once and never re-read: non-temporal so the 1GB
            // stream doesn't evict the K/V working set from L2.
            __builtin_nontemporal_store(p, &attnG[(size_t)row * S_LEN + j]);
        }
    }

    // ---------------- Phase 3: O = P @ V (async double-buffered) ----
    // Each wave owns one 16-column output tile (8 waves * 16 = 128 = D).
    // V tiles (32x128 fp32) staged via GLOBAL_LOAD_ASYNC_TO_LDS_B128:
    // 1024 float4 chunks per tile, 4 per thread (coalesced), 4 async ops/wave.
    const int nBase3 = wave * 16;

    auto issueTile = [&](int kt, int bufsel) {
        const int kBase = kt * 32;
        const uint32_t lbase = vsAddr[bufsel];
        #pragma unroll
        for (int j = 0; j < 4; ++j) {
            int c   = tid + 256 * j;          // float4 chunk id, lane-consecutive
            int row = c >> 5;                 // 32 float4 per 128-elem row
            int col = (c & 31) << 2;          // float column
            int goff = ((kBase + row) * D_DIM + col) * 4;          // global bytes
            uint32_t laddr = lbase + (uint32_t)((row * VSTRIDE + col) * 4);
            async_load_b128(laddr, goff, vG);
        }
    };

    issueTile(0, 0);
    issueTile(1, 1);

    v8f o = {};
    for (int kt = 0; kt < S_LEN / 32; ++kt) {
        // Retire tile kt's 4 per-wave async ops (tile kt+1 may stay in flight).
        if (kt < S_LEN / 32 - 1) async_wait_le4();
        else                     async_wait_le0();
        __syncthreads();                       // staged data visible to all waves

        const float* vbuf = (kt & 1) ? vs1 : vs0;
        const int kBase = kt * 32;

        v16bf a, b;
        #pragma unroll
        for (int vg = 0; vg < 8; ++vg) {
            int kk = kBase + ((vg >> 2) << 4) + half * 8 + ((vg & 3) << 1);
            a[2 * vg]     = (__bf16)sc[l16 * SROW + kk];
            a[2 * vg + 1] = (__bf16)sc[l16 * SROW + kk + 1];
        }
        const float* vcol = vbuf + nBase3 + l16;
        #pragma unroll
        for (int vg = 0; vg < 8; ++vg) {
            int kk = ((vg >> 2) << 4) + half * 8 + ((vg & 3) << 1);
            b[2 * vg]     = (__bf16)vcol[kk * VSTRIDE];
            b[2 * vg + 1] = (__bf16)vcol[(kk + 1) * VSTRIDE];
        }
        o = __builtin_amdgcn_wmma_f32_16x16x32_bf16(false, a, false, b,
                                                    (short)0, o, false, false);

        __syncthreads();                       // all reads of this buffer done
        if (kt + 2 < S_LEN / 32) issueTile(kt + 2, kt & 1);  // refill consumed buf
    }

    #pragma unroll
    for (int r = 0; r < 8; ++r) {
        int m = half * 8 + r;
        __builtin_nontemporal_store(o[r], &outG[m * D_DIM + nBase3 + l16]);
    }
}

extern "C" void kernel_launch(void* const* d_in, const int* in_sizes, int n_in,
                              void* d_out, int out_size, void* d_ws, size_t ws_size,
                              hipStream_t stream) {
    const float* q = (const float*)d_in[0];
    const float* k = (const float*)d_in[1];
    const float* v = (const float*)d_in[2];

    const int B = 4, H = 16;
    float* out  = (float*)d_out;                              // [B,H,S,D] first
    float* attn = out + (size_t)B * H * S_LEN * D_DIM;        // then [B,H,S,S]

    (void)hipFuncSetAttribute((const void*)sdpa_fused_kernel,
                              hipFuncAttributeMaxDynamicSharedMemorySize,
                              SMEM_BYTES);

    const int grid = B * H * (S_LEN / QT);   // 8192 workgroups
    sdpa_fused_kernel<<<grid, 256, SMEM_BYTES, stream>>>(q, k, v, out, attn);
}
